// GGNN_LCG_15839839387877
// MI455X (gfx1250) — matrix operand order repaired and checked by reading, not compiled, not run on hardware
//
#include <hip/hip_runtime.h>
#include <hip/hip_bf16.h>
#include <math.h>

// ---------------------------------------------------------------------------
// GGNN LCG message passing for MI455X (gfx1250, wave32, WMMA).
// GEMMs: bf16 inputs, f32 accumulate via v_wmma_f32_16x16x32_bf16.
// ---------------------------------------------------------------------------

typedef __attribute__((ext_vector_type(16))) __bf16 bf16x16;
typedef __attribute__((ext_vector_type(8)))  __bf16 bf16x8;
typedef __attribute__((ext_vector_type(4)))  __bf16 bf16x4;
typedef __attribute__((ext_vector_type(8)))  float  f32x8;
typedef __attribute__((ext_vector_type(4)))  float  f32x4;

// ------------------------- small utility kernels ---------------------------

__global__ void k_f32_to_bf16(const float* __restrict__ in, __bf16* __restrict__ out, int n) {
  int i = blockIdx.x * 256 + threadIdx.x;
  if (i < n) out[i] = (__bf16)in[i];
}

__global__ void k_zero_f32v4(float* __restrict__ p, int n4) {
  int i = blockIdx.x * 256 + threadIdx.x;
  if (i < n4) ((f32x4*)p)[i] = (f32x4){0.0f, 0.0f, 0.0f, 0.0f};
}

// aggr[dst[e]] += msg[src[e]] : one wave per edge, b128 gather + 4 f32 atomics
__global__ void k_scatter_add(const float* __restrict__ msg,
                              const int* __restrict__ src_idx,
                              const int* __restrict__ dst_idx,
                              float* __restrict__ aggr, int E) {
  int e = blockIdx.x * 8 + (threadIdx.x >> 5);   // 8 edges per 256-thread block
  if (e >= E) return;
  int lane = threadIdx.x & 31;
  int s = src_idx[e], d = dst_idx[e];
  f32x4 v = *(const f32x4*)(msg + (size_t)s * 128 + lane * 4);
  float* p = aggr + (size_t)d * 128 + lane * 4;
  __hip_atomic_fetch_add(p + 0, v[0], __ATOMIC_RELAXED, __HIP_MEMORY_SCOPE_AGENT);
  __hip_atomic_fetch_add(p + 1, v[1], __ATOMIC_RELAXED, __HIP_MEMORY_SCOPE_AGENT);
  __hip_atomic_fetch_add(p + 2, v[2], __ATOMIC_RELAXED, __HIP_MEMORY_SCOPE_AGENT);
  __hip_atomic_fetch_add(p + 3, v[3], __ATOMIC_RELAXED, __HIP_MEMORY_SCOPE_AGENT);
}

// GRU pointwise: gi/gh are (N,384) gate pre-activations, h is (N,128)
__global__ void k_gru_pointwise(const float* __restrict__ gi, const float* __restrict__ gh,
                                const float* __restrict__ h, float* __restrict__ hout, int N) {
  int g = blockIdx.x * 256 + threadIdx.x;          // L*128 = 7.68M
  if (g >= N * 128) return;
  int row = g >> 7, k = g & 127;
  size_t b = (size_t)row * 384;
  float ir = gi[b + k], iz = gi[b + 128 + k], in_ = gi[b + 256 + k];
  float hr = gh[b + k], hz = gh[b + 128 + k], hn  = gh[b + 256 + k];
  float r = 1.0f / (1.0f + __expf(-(ir + hr)));
  float z = 1.0f / (1.0f + __expf(-(iz + hz)));
  float n = tanhf(in_ + r * hn);
  hout[g] = (1.0f - z) * n + z * h[g];
}

// ------------------------------ WMMA GEMM ----------------------------------
// Y[N,M] = act(X[N,K] @ W[M,K]^T + bias), X fp32 (converted to bf16 in LDS),
// W pre-converted bf16 row-major (M,K): column m of a WMMA B tile is a
// contiguous K-run of W row m.
// Block: 256 threads = 8 waves; 64-row tile; wave w: row-tile (w&3),
// column tiles (w>>2), (w>>2)+2, ...; A-frags register-resident across tiles.
// Two independent WMMA accumulator chains, both starting from inline C=0;
// bias applied in the epilogue so no load gates the first WMMA.
// xorRow: read X row (r ^ xorRow) (l2l pair swap). CONCAT: X = [Xa | X2] along K.
template <int ACT, int K, bool CONCAT>
__launch_bounds__(256)
__global__ void k_gemm_bias(const float* __restrict__ X, const float* __restrict__ X2,
                            const __bf16* __restrict__ W, const float* __restrict__ bias,
                            float* __restrict__ Y, int N, int M, int xorRow) {
  constexpr int LDK = K + 8;        // pad: keeps rows 16B-aligned, breaks bank conflicts
  constexpr int NK  = K / 32;       // k-steps of 32 (bf16 WMMA depth); even (4 or 8)
  __shared__ __bf16 Xs[64 * LDK];

  const int tid = threadIdx.x;
  const int n0  = blockIdx.x * 64;

  // ---- stage 64 x K tile of X as bf16 into LDS (float4-vectorized) ----
  constexpr int KC = K / 4;
  for (int idx = tid; idx < 64 * KC; idx += 256) {
    int r  = idx / KC;
    int kc = (idx - r * KC) * 4;
    int grow = n0 + r;
    f32x4 v = (f32x4){0.0f, 0.0f, 0.0f, 0.0f};
    if (grow < N) {
      const float* xp;
      if (CONCAT)   // [Xa | X2], each (N,128); chunk never straddles the split
        xp = (kc < 128) ? (X + (size_t)grow * 128 + kc)
                        : (X2 + (size_t)grow * 128 + (kc - 128));
      else
        xp = X + (size_t)(grow ^ xorRow) * K + kc;
      v = *(const f32x4*)xp;
    }
    bf16x4 bv;
    bv[0] = (__bf16)v[0]; bv[1] = (__bf16)v[1];
    bv[2] = (__bf16)v[2]; bv[3] = (__bf16)v[3];
    *(bf16x4*)&Xs[r * LDK + kc] = bv;
  }
  __syncthreads();

  const int wave = tid >> 5;
  const int lane = tid & 31;
  const int half = lane >> 4;       // 0: lanes 0-15, 1: lanes 16-31
  const int lr   = lane & 15;
  const int rt   = wave & 3;        // row tile within block (16 rows each)
  const int cg   = wave >> 2;       // column-tile phase (stride 2)

  // ---- load all A fragments for this wave's 16-row strip ----
  // A 16x32 frag in this code's contiguous k order (matches B): lanes 0-15 hold K 0-15, lanes 16-31 hold
  // K 16-31 of the same row.
  bf16x16 afrag[NK];
  {
    const __bf16* arow = &Xs[(rt * 16 + lr) * LDK + half * 16];
#pragma unroll
    for (int kk = 0; kk < NK; ++kk) {
      bf16x8 lo = *(const bf16x8*)(arow + kk * 32);
      bf16x8 hi = *(const bf16x8*)(arow + kk * 32 + 8);
      bf16x16 a;
#pragma unroll
      for (int j = 0; j < 8; ++j) { a[j] = lo[j]; a[j + 8] = hi[j]; }
      afrag[kk] = a;
    }
  }

  const bool full = (n0 + 64) <= N;   // uniform: interior block -> no store guards

  // ---- sweep column tiles; B frag: lane holds one output column (= W row) ----
  const int nmt = M >> 4;
  for (int mt = cg; mt < nmt; mt += 2) {
    const float bv = bias[mt * 16 + lr];   // completes in the WMMA shadow
    f32x8 acc0 = {}, acc1 = {};            // inline C=0 for both chains

    // B 32x16 frag in this code's contiguous k order: lanes 0-15 hold K 0-15 of their column, lanes 16-31 K 16-31.
    const __bf16* wp = W + (size_t)(mt * 16 + lr) * K + half * 16;
    __builtin_prefetch(wp + (size_t)32 * K, 0, 1);   // next column tile (mt+2)
#pragma unroll
    for (int kk = 0; kk < NK; kk += 2) {             // two independent WMMA chains
      bf16x16 b0 = *(const bf16x16*)(wp + kk * 32);
      bf16x16 b1 = *(const bf16x16*)(wp + (kk + 1) * 32);
      acc0 = __builtin_amdgcn_wmma_f32_16x16x32_bf16(false, afrag[kk], false, b0,
                                                     (short)0, acc0, false, false);
      acc1 = __builtin_amdgcn_wmma_f32_16x16x32_bf16(false, afrag[kk + 1], false, b1,
                                                     (short)0, acc1, false, false);
    }

    // C/D layout: VGPR i -> row i (lanes 0-15) / row i+8 (lanes 16-31), col lr.
    float* yp = Y + (size_t)(n0 + rt * 16 + half * 8) * M + mt * 16 + lr;
    if (full) {
#pragma unroll
      for (int i = 0; i < 8; ++i) {
        float v = acc0[i] + acc1[i] + bv;
        if (ACT == 1) v = v > 0.0f ? v : 0.0f;
        yp[(size_t)i * M] = v;
      }
    } else {
#pragma unroll
      for (int i = 0; i < 8; ++i) {
        float v = acc0[i] + acc1[i] + bv;
        if (ACT == 1) v = v > 0.0f ? v : 0.0f;
        int grow = n0 + rt * 16 + half * 8 + i;
        if (grow < N) yp[(size_t)i * M] = v;
      }
    }
  }
}

// ------------------------------- host side ---------------------------------

extern "C" void kernel_launch(void* const* d_in, const int* in_sizes, int n_in,
                              void* d_out, int out_size, void* d_ws, size_t ws_size,
                              hipStream_t stream) {
  (void)n_in; (void)out_size; (void)ws_size;

  const int*   l_edge = (const int*)d_in[2];
  const int*   c_edge = (const int*)d_in[3];
  const float* l_emb0 = (const float*)d_in[4];
  const float* c_emb0 = (const float*)d_in[5];
  const float* W_l2c0 = (const float*)d_in[6];
  const float* b_l2c0 = (const float*)d_in[7];
  const float* W_l2c1 = (const float*)d_in[8];
  const float* b_l2c1 = (const float*)d_in[9];
  const float* W_c2l0 = (const float*)d_in[10];
  const float* b_c2l0 = (const float*)d_in[11];
  const float* W_c2l1 = (const float*)d_in[12];
  const float* b_c2l1 = (const float*)d_in[13];
  const float* W_l2l0 = (const float*)d_in[14];
  const float* b_l2l0 = (const float*)d_in[15];
  const float* W_l2l1 = (const float*)d_in[16];
  const float* b_l2l1 = (const float*)d_in[17];
  const float* W_cih  = (const float*)d_in[18];
  const float* W_chh  = (const float*)d_in[19];
  const float* b_cih  = (const float*)d_in[20];
  const float* b_chh  = (const float*)d_in[21];
  const float* W_lih  = (const float*)d_in[22];
  const float* W_lhh  = (const float*)d_in[23];
  const float* b_lih  = (const float*)d_in[24];
  const float* b_lhh  = (const float*)d_in[25];

  const int E = in_sizes[2];
  const int L = in_sizes[4] / 128;
  const int C = in_sizes[5] / 128;
  const int N_ITER = 8;

  float* out_l = (float*)d_out;                          // (9, L, 128)
  float* out_c = out_l + (size_t)(N_ITER + 1) * L * 128; // (9, C, 128)

  // bump allocator on workspace, 256B aligned
  char*  wsb = (char*)d_ws;
  size_t off = 0;
  auto alloc = [&](size_t bytes) -> char* {
    off = (off + 255) & ~(size_t)255;
    char* p = wsb + off;
    off += bytes;
    return p;
  };

  // bf16 weight copies (tiny; L2 resident)
  __bf16* wb_l2c0 = (__bf16*)alloc((size_t)128 * 128 * 2);
  __bf16* wb_l2c1 = (__bf16*)alloc((size_t)128 * 128 * 2);
  __bf16* wb_c2l0 = (__bf16*)alloc((size_t)128 * 128 * 2);
  __bf16* wb_c2l1 = (__bf16*)alloc((size_t)128 * 128 * 2);
  __bf16* wb_l2l0 = (__bf16*)alloc((size_t)128 * 128 * 2);
  __bf16* wb_l2l1 = (__bf16*)alloc((size_t)128 * 128 * 2);
  __bf16* wb_cih  = (__bf16*)alloc((size_t)384 * 128 * 2);
  __bf16* wb_chh  = (__bf16*)alloc((size_t)384 * 128 * 2);
  __bf16* wb_lih  = (__bf16*)alloc((size_t)384 * 256 * 2);
  __bf16* wb_lhh  = (__bf16*)alloc((size_t)384 * 128 * 2);

  // fp32 activation buffers
  float* hidden  = (float*)alloc((size_t)L * 128 * 4);
  float* msg_l2c = (float*)alloc((size_t)L * 128 * 4);
  float* msg_c2l = (float*)alloc((size_t)C * 128 * 4);
  float* msg_l2l = (float*)alloc((size_t)L * 128 * 4);
  float* aggr_c  = (float*)alloc((size_t)C * 128 * 4);
  float* aggr_l  = (float*)alloc((size_t)L * 128 * 4);
  float* gi      = (float*)alloc((size_t)L * 384 * 4);  // shared by c/l GRUs
  float* gh      = (float*)alloc((size_t)L * 384 * 4);

  auto cvt = [&](const float* src, __bf16* dst, int n) {
    k_f32_to_bf16<<<(n + 255) / 256, 256, 0, stream>>>(src, dst, n);
  };
  cvt(W_l2c0, wb_l2c0, 128 * 128);
  cvt(W_l2c1, wb_l2c1, 128 * 128);
  cvt(W_c2l0, wb_c2l0, 128 * 128);
  cvt(W_c2l1, wb_c2l1, 128 * 128);
  cvt(W_l2l0, wb_l2l0, 128 * 128);
  cvt(W_l2l1, wb_l2l1, 128 * 128);
  cvt(W_cih, wb_cih, 384 * 128);
  cvt(W_chh, wb_chh, 384 * 128);
  cvt(W_lih, wb_lih, 384 * 256);
  cvt(W_lhh, wb_lhh, 384 * 128);

  auto gemm128 = [&](const float* X, const __bf16* Wb, const float* bias, float* Y,
                     int N, int M, int xr, int relu) {
    dim3 grid((N + 63) / 64);
    if (relu) k_gemm_bias<1, 128, false><<<grid, 256, 0, stream>>>(X, nullptr, Wb, bias, Y, N, M, xr);
    else      k_gemm_bias<0, 128, false><<<grid, 256, 0, stream>>>(X, nullptr, Wb, bias, Y, N, M, xr);
  };
  auto gemm256cat = [&](const float* Xa, const float* Xb, const __bf16* Wb,
                        const float* bias, float* Y, int N, int M) {
    dim3 grid((N + 63) / 64);
    k_gemm_bias<0, 256, true><<<grid, 256, 0, stream>>>(Xa, Xb, Wb, bias, Y, N, M, 0);
  };
  auto zero = [&](float* p, size_t n) {
    k_zero_f32v4<<<(unsigned)((n / 4 + 255) / 256), 256, 0, stream>>>(p, (int)(n / 4));
  };

  // iteration 0 snapshots
  hipMemcpyAsync(out_l, l_emb0, (size_t)L * 128 * 4, hipMemcpyDeviceToDevice, stream);
  hipMemcpyAsync(out_c, c_emb0, (size_t)C * 128 * 4, hipMemcpyDeviceToDevice, stream);

  for (int t = 0; t < N_ITER; ++t) {
    const float* lcur = out_l + (size_t)t * L * 128;
    const float* ccur = out_c + (size_t)t * C * 128;
    float* lnext = out_l + (size_t)(t + 1) * L * 128;
    float* cnext = out_c + (size_t)(t + 1) * C * 128;

    // l2c MLP over literal nodes
    gemm128(lcur, wb_l2c0, b_l2c0, hidden, L, 128, 0, 1);
    gemm128(hidden, wb_l2c1, b_l2c1, msg_l2c, L, 128, 0, 0);
    // c2l MLP over clause nodes
    gemm128(ccur, wb_c2l0, b_c2l0, hidden, C, 128, 0, 1);
    gemm128(hidden, wb_c2l1, b_c2l1, msg_c2l, C, 128, 0, 0);
    // l2l MLP with pair swap (row r reads l_emb[r^1])
    gemm128(lcur, wb_l2l0, b_l2l0, hidden, L, 128, 1, 1);
    gemm128(hidden, wb_l2l1, b_l2l1, msg_l2l, L, 128, 0, 0);

    // clause aggregation: aggr_c[c_edge[e]] += msg_l2c[l_edge[e]]
    zero(aggr_c, (size_t)C * 128);
    k_scatter_add<<<(E + 7) / 8, 256, 0, stream>>>(msg_l2c, l_edge, c_edge, aggr_c, E);

    // clause GRU
    gemm128(aggr_c, wb_cih, b_cih, gi, C, 384, 0, 0);
    gemm128(ccur, wb_chh, b_chh, gh, C, 384, 0, 0);
    k_gru_pointwise<<<((size_t)C * 128 + 255) / 256, 256, 0, stream>>>(gi, gh, ccur, cnext, C);

    // literal aggregation: aggr_l[l_edge[e]] += msg_c2l[c_edge[e]]
    zero(aggr_l, (size_t)L * 128);
    k_scatter_add<<<(E + 7) / 8, 256, 0, stream>>>(msg_c2l, c_edge, l_edge, aggr_l, E);

    // literal GRU (input = [c2l_aggr | l2l_msg] fused into staging, K = 256)
    gemm256cat(aggr_l, msg_l2l, wb_lih, b_lih, gi, L, 384);
    gemm128(lcur, wb_lhh, b_lhh, gh, L, 384, 0, 0);
    k_gru_pointwise<<<((size_t)L * 128 + 255) / 256, 256, 0, stream>>>(gi, gh, lcur, lnext, L);
  }
}
